// DeepseekV4Attention_18425409700456
// MI455X (gfx1250) — matrix operand order repaired and checked by reading, not compiled, not run on hardware
//
#include <hip/hip_runtime.h>
#include <math.h>

typedef __attribute__((ext_vector_type(16))) _Float16 v16h;
typedef __attribute__((ext_vector_type(8)))  _Float16 v8h;
typedef __attribute__((ext_vector_type(8)))  float    v8f;

#define BB     2
#define SS     4096
#define HIDC   2048
#define NHC    16
#define HDC    128
#define NWC    1024
#define RATIOC 4
#define MTOT   (BB * SS)          // 8192 rows

// ---------------------------------------------------------------------------
// WMMA helpers (gfx1250 wave32: D = A(16x32 f16) * B(32x16 f16) + C(16x16 f32))
// ---------------------------------------------------------------------------
__device__ __forceinline__ v8f wmma16(v16h a, v16h b, v8f c) {
  return __builtin_amdgcn_wmma_f32_16x16x32_f16(false, a, false, b, (short)0, c,
                                                false, false);
}

// A-operand: lane<16 holds row (m), K halves {0..7,16..23}; lane>=16 same row,
// K halves {8..15,24..31}. rowp = per-lane pointer to that row's 32-half chunk.
__device__ __forceinline__ v16h load_a_frag(const _Float16* rowp, int lane) {
  const int o = (lane < 16) ? 0 : 8;
  v8h lo = *(const v8h*)(rowp + o);
  v8h hi = *(const v8h*)(rowp + o + 16);
  return __builtin_shufflevector(lo, hi, 0, 1, 2, 3, 4, 5, 6, 7, 8, 9, 10, 11,
                                 12, 13, 14, 15);
}

// B-operand: lane holds column n = lane%16; lane<16 K={0..7,16..23}, lane>=16
// K={8..15,24..31}. colp = per-lane pointer to that column's 32 contiguous K halves.
__device__ __forceinline__ v16h load_b_frag(const _Float16* colp, int lane) {
  const int o = (lane < 16) ? 0 : 8;
  v8h lo = *(const v8h*)(colp + o);
  v8h hi = *(const v8h*)(colp + o + 16);
  return __builtin_shufflevector(lo, hi, 0, 1, 2, 3, 4, 5, 6, 7, 8, 9, 10, 11,
                                 12, 13, 14, 15);
}

// ---------------------------------------------------------------------------
// Elementwise prep kernels
// ---------------------------------------------------------------------------
__global__ void k_f32_to_f16(const float* __restrict__ x,
                             _Float16* __restrict__ y, int n) {
  int i = blockIdx.x * blockDim.x + threadIdx.x;
  if (i < n) y[i] = (_Float16)x[i];
}

// W[k][n] f32 -> WT[n][k] f16
__global__ void k_transpose_f16(const float* __restrict__ w,
                                _Float16* __restrict__ wt, int K, int N) {
  int i = blockIdx.x * blockDim.x + threadIdx.x;
  if (i < K * N) {
    int k = i / N, n = i - k * N;
    wt[(size_t)n * K + k] = (_Float16)w[i];
  }
}

// ---------------------------------------------------------------------------
// Tiled WMMA GEMM: C[M,N] = A16[M,K] @ (BT16[N,K])^T.  One wave per 32x64 tile.
// Register double-buffered: prefetch K-chunk ki+1 fragments into the alternate
// register set while WMMAs consume chunk ki (kills load->wmma waits and the
// WAR-hazard v_nops on the B operand registers).
// ---------------------------------------------------------------------------
template <typename OutT>
__global__ void k_gemm_wmma(const _Float16* __restrict__ A,
                            const _Float16* __restrict__ BT,
                            OutT* __restrict__ C, int M, int N, int K) {
  const int lane = threadIdx.x;
  const int lm = lane & 15;
  const int m0 = blockIdx.x * 32;
  const int n0 = blockIdx.y * 64;

  v8f acc[2][4];
#pragma unroll
  for (int mt = 0; mt < 2; ++mt)
#pragma unroll
    for (int nt = 0; nt < 4; ++nt)
#pragma unroll
      for (int v = 0; v < 8; ++v) acc[mt][nt][v] = 0.0f;

  const _Float16* a0p = A + (size_t)(m0 + lm) * K;
  const _Float16* a1p = A + (size_t)(m0 + 16 + lm) * K;
  const _Float16* bp[4];
#pragma unroll
  for (int nt = 0; nt < 4; ++nt)
    bp[nt] = BT + (size_t)(n0 + nt * 16 + lm) * K;

  // double-buffered fragment registers
  v16h a0[2], a1[2], bf[2][4];
  a0[0] = load_a_frag(a0p, lane);
  a1[0] = load_a_frag(a1p, lane);
#pragma unroll
  for (int nt = 0; nt < 4; ++nt) bf[0][nt] = load_b_frag(bp[nt], lane);

  const int nk = K >> 5;  // K / 32
#pragma unroll 2
  for (int ki = 0; ki < nk; ++ki) {
    const int cur = ki & 1;
    const int nxt = cur ^ 1;
    // clamped prefetch: last iteration re-fetches same chunk (no OOB)
    const int kn = (ki + 1 < nk) ? (ki + 1) << 5 : ki << 5;
    a0[nxt] = load_a_frag(a0p + kn, lane);
    a1[nxt] = load_a_frag(a1p + kn, lane);
#pragma unroll
    for (int nt = 0; nt < 4; ++nt) bf[nxt][nt] = load_b_frag(bp[nt] + kn, lane);
#pragma unroll
    for (int nt = 0; nt < 4; ++nt) {
      acc[0][nt] = wmma16(a0[cur], bf[cur][nt], acc[0][nt]);
      acc[1][nt] = wmma16(a1[cur], bf[cur][nt], acc[1][nt]);
    }
  }

  const int rbase = (lane < 16) ? 0 : 8;  // C layout: row = v + 8*(lane>=16)
#pragma unroll
  for (int mt = 0; mt < 2; ++mt)
#pragma unroll
    for (int nt = 0; nt < 4; ++nt)
#pragma unroll
      for (int v = 0; v < 8; ++v) {
        int row = m0 + mt * 16 + rbase + v;
        int col = n0 + nt * 16 + lm;
        C[(size_t)row * N + col] = (OutT)acc[mt][nt][v];
      }
}

// ---------------------------------------------------------------------------
// Partial interleaved RoPE on q (in place).  q layout [b][s][h*128+d] f16.
// Thread per (b,s,h,pair j in 0..31), pair dims = 64+2j, 64+2j+1, pos = s.
// ---------------------------------------------------------------------------
__global__ void k_rope_q(_Float16* __restrict__ q) {
  const int total = BB * SS * NHC * 32;
  int i = blockIdx.x * blockDim.x + threadIdx.x;
  if (i >= total) return;
  int j = i & 31;
  int t = i >> 5;
  int h = t & (NHC - 1);
  int bs = t >> 4;
  int pos = bs & (SS - 1);
  size_t base = (size_t)bs * (NHC * HDC) + (size_t)h * HDC + 64 + 2 * j;
  float a = (float)q[base], b = (float)q[base + 1];
  float invf = powf(10000.0f, -(float)j * (1.0f / 32.0f));
  float ang = (float)pos * invf;
  float c = cosf(ang), s = sinf(ang);
  q[base] = (_Float16)(a * c - b * s);
  q[base + 1] = (_Float16)(a * s + b * c);
}

// ---------------------------------------------------------------------------
// Pooling + key RoPE.  Block = (b, window w), 128 threads = feature d.
// Slots 0..3: prev window positions, features [0:128); slots 4..7: current
// window positions, features [128:256).  Per-feature softmax over 8 slots.
// Writes k_rot[b][w][128] (RoPE'd, pos = 4w) and vT[b][d][w] (pooled).
// ---------------------------------------------------------------------------
__global__ void k_pool(const float* __restrict__ kv,
                       const float* __restrict__ gate,
                       const float* __restrict__ ape,
                       _Float16* __restrict__ krot, _Float16* __restrict__ vT) {
  const int w = blockIdx.x & (NWC - 1);
  const int b = blockIdx.x >> 10;
  const int d = threadIdx.x;  // 0..127

  float kvs[8], gs[8];
#pragma unroll
  for (int j = 0; j < 4; ++j) {
    if (w == 0) {
      kvs[j] = 0.0f;
      gs[j] = -1e30f;
    } else {
      size_t idx = ((size_t)b * SS + (size_t)(w - 1) * RATIOC + j) * 256 + d;
      kvs[j] = kv[idx];
      gs[j] = gate[idx] + ape[j * 256 + d];
    }
    size_t idx2 = ((size_t)b * SS + (size_t)w * RATIOC + j) * 256 + HDC + d;
    kvs[4 + j] = kv[idx2];
    gs[4 + j] = gate[idx2] + ape[j * 256 + HDC + d];
  }
  float mx = gs[0];
#pragma unroll
  for (int j = 1; j < 8; ++j) mx = fmaxf(mx, gs[j]);
  float sum = 0.0f, accv = 0.0f;
#pragma unroll
  for (int j = 0; j < 8; ++j) {
    float e = expf(gs[j] - mx);
    sum += e;
    accv += e * kvs[j];
  }
  float pooled = accv / sum;

  __shared__ float pl[HDC];
  pl[d] = pooled;
  __syncthreads();

  float outv = pooled;
  if (d >= 64) {
    int j = (d - 64) >> 1;
    float invf = powf(10000.0f, -(float)j * (1.0f / 32.0f));
    float ang = (float)(w * RATIOC) * invf;
    float c = cosf(ang), s = sinf(ang);
    float a = pl[64 + 2 * j], bb = pl[64 + 2 * j + 1];
    outv = ((d & 1) == 0) ? (a * c - bb * s) : (a * s + bb * c);
  }
  krot[((size_t)b * NWC + w) * HDC + d] = (_Float16)outv;
  vT[((size_t)b * HDC + d) * NWC + w] = (_Float16)pooled;
}

// ---------------------------------------------------------------------------
// Flash attention with sink.  One wave per (b, h, 16 query rows).
// Online softmax state initialized from the sink logit: m=sinks[h], l=1, acc=0
// (exactly reproduces softmax-with-sink-column-then-drop).
// Causal mask: query q attends window k iff q >= 4k+3.
// ---------------------------------------------------------------------------
__global__ void k_attn(const _Float16* __restrict__ q16,
                       const _Float16* __restrict__ krot,
                       const _Float16* __restrict__ vT,
                       const float* __restrict__ sinks,
                       _Float16* __restrict__ attn16) {
  const int lane = threadIdx.x;
  const int lm = lane & 15;
  const int rbase = (lane < 16) ? 0 : 8;
  const int qt = blockIdx.x;
  const int h = blockIdx.y;
  const int b = blockIdx.z;
  const int q0 = qt * 16;

  __shared__ float Sld[16][32];
  __shared__ _Float16 Pld[16][32];
  __shared__ float rowScale[16];
  __shared__ float rowL[16];

  // Q fragments: 4 K-chunks of 32 features each (A layout)
  const _Float16* qrow =
      q16 + ((size_t)(b * SS + q0 + lm)) * (NHC * HDC) + (size_t)h * HDC;
  v16h qa[4];
#pragma unroll
  for (int c = 0; c < 4; ++c) qa[c] = load_a_frag(qrow + c * 32, lane);

  v8f acc[8];
#pragma unroll
  for (int nt = 0; nt < 8; ++nt)
#pragma unroll
    for (int v = 0; v < 8; ++v) acc[nt][v] = 0.0f;

  float m_r = sinks[h];  // meaningful on lanes 0..15 (row owners)
  float l_r = 1.0f;
  const float scale = 0.08838834764831845f;  // 1/sqrt(128)

  const int khi = (q0 + 12) >> 2;   // last key visible to row q0+15
  const int nkb = (khi + 32) >> 5;  // ceil((khi+1)/32) key blocks of 32

  const _Float16* krow = krot + (size_t)(b * NWC + lm) * HDC;
  const _Float16* vrow = vT + (size_t)(b * HDC + lm) * NWC;

  for (int kb = 0; kb < nkb; ++kb) {
    const int kbase = kb * 32;

    // --- scores S = Q @ K^T (16 rows x 32 keys): batch all 8 B frags first
    v16h bk[8];
#pragma unroll
    for (int t = 0; t < 2; ++t)
#pragma unroll
      for (int c = 0; c < 4; ++c)
        bk[t * 4 + c] =
            load_b_frag(krow + (size_t)(kbase + t * 16) * HDC + c * 32, lane);

    v8f sa[2];
#pragma unroll
    for (int t = 0; t < 2; ++t)
#pragma unroll
      for (int v = 0; v < 8; ++v) sa[t][v] = 0.0f;
#pragma unroll
    for (int c = 0; c < 4; ++c) {
      sa[0] = wmma16(qa[c], bk[c], sa[0]);
      sa[1] = wmma16(qa[c], bk[4 + c], sa[1]);
    }

    // stage C-layout scores to LDS
#pragma unroll
    for (int t = 0; t < 2; ++t)
#pragma unroll
      for (int v = 0; v < 8; ++v) Sld[rbase + v][t * 16 + lm] = sa[t][v];
    __syncthreads();

    // row-wise masked online softmax (lane r owns row r)
    if (lane < 16) {
      const int qr = q0 + lane;
      float mx = m_r;
#pragma unroll
      for (int j = 0; j < 32; ++j) {
        float s = Sld[lane][j] * scale;
        if (qr < 4 * (kbase + j) + 3) s = -1e30f;
        mx = fmaxf(mx, s);
      }
      float f = expf(m_r - mx);
      float sum = 0.0f;
#pragma unroll
      for (int j = 0; j < 32; ++j) {
        float s = Sld[lane][j] * scale;
        if (qr < 4 * (kbase + j) + 3) s = -1e30f;
        float p = expf(s - mx);
        sum += p;
        Pld[lane][j] = (_Float16)p;
      }
      l_r = l_r * f + sum;
      m_r = mx;
      rowScale[lane] = f;
    }
    __syncthreads();

    // --- rescale accumulators, then acc += P @ V: batch 8 V frags first
    v16h vb[8];
#pragma unroll
    for (int nt = 0; nt < 8; ++nt)
      vb[nt] = load_b_frag(vrow + (size_t)(nt * 16) * NWC + kbase, lane);

    float fs[8];
#pragma unroll
    for (int v = 0; v < 8; ++v) fs[v] = rowScale[rbase + v];
#pragma unroll
    for (int nt = 0; nt < 8; ++nt)
#pragma unroll
      for (int v = 0; v < 8; ++v) acc[nt][v] *= fs[v];

    v16h pa = load_a_frag(&Pld[lm][0], lane);
#pragma unroll
    for (int nt = 0; nt < 8; ++nt) acc[nt] = wmma16(pa, vb[nt], acc[nt]);
    __syncthreads();
  }

  if (lane < 16) rowL[lane] = l_r;
  __syncthreads();
  float inv[8];
#pragma unroll
  for (int v = 0; v < 8; ++v) inv[v] = 1.0f / rowL[rbase + v];
#pragma unroll
  for (int nt = 0; nt < 8; ++nt)
#pragma unroll
    for (int v = 0; v < 8; ++v) {
      size_t o = ((size_t)(b * SS + q0 + rbase + v)) * (NHC * HDC) +
                 (size_t)h * HDC + nt * 16 + lm;
      attn16[o] = (_Float16)(acc[nt][v] * inv[v]);
    }
}

// ---------------------------------------------------------------------------
// Host launcher
// ---------------------------------------------------------------------------
extern "C" void kernel_launch(void* const* d_in, const int* in_sizes, int n_in,
                              void* d_out, int out_size, void* d_ws,
                              size_t ws_size, hipStream_t stream) {
  (void)in_sizes; (void)n_in; (void)out_size; (void)ws_size;
  const float* hidden = (const float*)d_in[0];
  const float* wq = (const float*)d_in[1];
  const float* wkv = (const float*)d_in[2];
  const float* wgate = (const float*)d_in[3];
  const float* ape = (const float*)d_in[4];
  const float* sinks = (const float*)d_in[5];
  const float* wo = (const float*)d_in[6];
  float* out = (float*)d_out;

  char* ws = (char*)d_ws;
  size_t off = 0;
  auto carve = [&](size_t nbytes) {
    char* p = ws + off;
    off += (nbytes + 255) & ~(size_t)255;
    return (void*)p;
  };

  _Float16* hidden16 = (_Float16*)carve((size_t)MTOT * HIDC * 2);      // 32 MB
  _Float16* q16 = (_Float16*)carve((size_t)MTOT * 2048 * 2);           // 32 MB
  _Float16* attn16 = (_Float16*)carve((size_t)MTOT * 2048 * 2);        // 32 MB
  _Float16* wqT = (_Float16*)carve((size_t)2048 * 2048 * 2);           // 8 MB
  _Float16* woT = (_Float16*)carve((size_t)2048 * 2048 * 2);           // 8 MB
  float* kv32 = (float*)carve((size_t)MTOT * 256 * 4);                 // 8 MB
  float* gate32 = (float*)carve((size_t)MTOT * 256 * 4);               // 8 MB
  _Float16* wkvT = (_Float16*)carve((size_t)256 * 2048 * 2);           // 1 MB
  _Float16* wgateT = (_Float16*)carve((size_t)256 * 2048 * 2);         // 1 MB
  _Float16* krot = (_Float16*)carve((size_t)BB * NWC * HDC * 2);       // .5 MB
  _Float16* vT = (_Float16*)carve((size_t)BB * HDC * NWC * 2);         // .5 MB

  // 1) hidden -> f16
  {
    int n = MTOT * HIDC;
    k_f32_to_f16<<<(n + 255) / 256, 256, 0, stream>>>(hidden, hidden16, n);
  }
  // 2) weight transposes (+ f16 convert)
  k_transpose_f16<<<(2048 * 2048 + 255) / 256, 256, 0, stream>>>(wq, wqT, 2048, 2048);
  k_transpose_f16<<<(2048 * 256 + 255) / 256, 256, 0, stream>>>(wkv, wkvT, 2048, 256);
  k_transpose_f16<<<(2048 * 256 + 255) / 256, 256, 0, stream>>>(wgate, wgateT, 2048, 256);
  k_transpose_f16<<<(2048 * 2048 + 255) / 256, 256, 0, stream>>>(wo, woT, 2048, 2048);

  // 3) projections (WMMA)
  k_gemm_wmma<_Float16><<<dim3(MTOT / 32, 2048 / 64), 32, 0, stream>>>(
      hidden16, wqT, q16, MTOT, 2048, 2048);
  k_gemm_wmma<float><<<dim3(MTOT / 32, 256 / 64), 32, 0, stream>>>(
      hidden16, wkvT, kv32, MTOT, 256, 2048);
  k_gemm_wmma<float><<<dim3(MTOT / 32, 256 / 64), 32, 0, stream>>>(
      hidden16, wgateT, gate32, MTOT, 256, 2048);

  // 4) RoPE on q, pooling + key RoPE
  {
    int n = BB * SS * NHC * 32;
    k_rope_q<<<(n + 255) / 256, 256, 0, stream>>>(q16);
  }
  k_pool<<<BB * NWC, HDC, 0, stream>>>(kv32, gate32, ape, krot, vT);

  // 5) flash attention with sink (WMMA)
  k_attn<<<dim3(SS / 16, NHC, BB), 32, 0, stream>>>(q16, krot, vT, sinks,
                                                    attn16);

  // 6) output projection (WMMA) -> d_out (f32)
  k_gemm_wmma<float><<<dim3(MTOT / 32, 2048 / 64), 32, 0, stream>>>(
      attn16, woT, out, MTOT, 2048, 2048);
}
